// GraphTransformerLayer_67267777789978
// MI455X (gfx1250) — matrix operand order repaired and checked by reading, not compiled, not run on hardware
//
#include <hip/hip_runtime.h>
#include <hip/hip_bf16.h>

#define BATCH 2048
#define NNB   32
#define DDIM  64
#define HH    8

typedef __attribute__((ext_vector_type(8)))  unsigned int u32x8;
typedef __attribute__((ext_vector_type(8)))  float        v8f;
typedef __attribute__((ext_vector_type(16))) __bf16       v16bf;

__device__ __forceinline__ unsigned short f2bf(float x) {
    union { float f; unsigned u; } v; v.f = x;
    unsigned r = v.u + 0x7FFFu + ((v.u >> 16) & 1u);
    return (unsigned short)(r >> 16);
}

__device__ __forceinline__ v8f wmma_bf16(u32x8 a, u32x8 b, v8f c) {
    return __builtin_amdgcn_wmma_f32_16x16x32_bf16(
        false, __builtin_bit_cast(v16bf, a),
        false, __builtin_bit_cast(v16bf, b),
        (short)0, c, false, false);
}

// ---------------- Kernel 0: prep (Wvsum = sum_h Wv rows; Wk -> bf16) -------
__global__ __launch_bounds__(256)
void prep_kernel(const float* __restrict__ Wv, const float* __restrict__ Wk,
                 float* __restrict__ Wvsum, unsigned short* __restrict__ Wk_bf) {
    int i = blockIdx.x * 256 + threadIdx.x;       // 32768 threads total
    if (i < DDIM * DDIM) {
        int ii = i >> 6, d = i & 63;
        float s = 0.f;
        #pragma unroll
        for (int h = 0; h < HH; ++h) s += Wv[(ii * HH + h) * DDIM + d];
        Wvsum[i] = s;
    }
    Wk_bf[i] = f2bf(Wk[i]);                        // 512*64 = 32768 elements
}

// ---------------- Kernel A: full attention per batch row -------------------
__global__ __launch_bounds__(256)
void attn_kernel(const float* __restrict__ center, const float* __restrict__ neighbor,
                 const float* __restrict__ Wq, const unsigned short* __restrict__ Wk_bf,
                 const float* __restrict__ Wvsum, const float* __restrict__ Wo,
                 const float* __restrict__ bo, float* __restrict__ h_pre) {
    __shared__ unsigned short s_xn[NNB * DDIM];          // bf16 neighbor rows (A matrix)
    __shared__ unsigned short s_kh[NNB * DDIM * HH];     // Kh: [n][j*8+h] bf16, 32 KB
    __shared__ unsigned short s_qh[DDIM * HH];           // Qh: [i*8+h] bf16 (pre-scaled 1/8)
    __shared__ float s_vsum[NNB * DDIM];                 // vsum[n][i]
    __shared__ float s_center[DDIM];
    __shared__ float s_attn[DDIM];

    const int b = blockIdx.x;
    const int tid = threadIdx.x;
    const int w = tid >> 5;
    const int lane = tid & 31;
    const int lh = lane >> 4;       // lane half
    const int lm = lane & 15;

    if (tid < DDIM) { s_center[tid] = center[b * DDIM + tid]; s_attn[tid] = 0.f; }
    for (int i = tid; i < NNB * DDIM; i += 256)
        s_xn[i] = f2bf(neighbor[b * NNB * DDIM + i]);
    __syncthreads();

    // ---- Q projection (VALU matvec), fold 1/sqrt(D) = 1/8 ----
    for (int c = tid; c < DDIM * HH; c += 256) {
        float acc = 0.f;
        #pragma unroll
        for (int d = 0; d < DDIM; ++d) acc += s_center[d] * Wq[c * DDIM + d];
        s_qh[c] = f2bf(acc * 0.125f);
    }

    // ---- A fragments of xn (32x64, K=d), reused for K-proj and Vsum-proj ----
    u32x8 axn[2][2];
    #pragma unroll
    for (int mt = 0; mt < 2; ++mt)
        #pragma unroll
        for (int ks = 0; ks < 2; ++ks) {
            const unsigned short* p = s_xn + (mt * 16 + lm) * DDIM + ks * 32 + lh * 16;
            uint4 lo = *(const uint4*)(p);
            uint4 hi = *(const uint4*)(p + 8);
            u32x8 t = {lo.x, lo.y, lo.z, lo.w, hi.x, hi.y, hi.z, hi.w};
            axn[mt][ks] = t;
        }

    // ---- K projection: C[n, j*8+h] = xn @ Wk^T ; wave w owns ntiles 4w..4w+3 ----
    #pragma unroll
    for (int t = 0; t < 4; ++t) {
        int nt = w * 4 + t;
        u32x8 bk[2];
        #pragma unroll
        for (int ks = 0; ks < 2; ++ks) {
            const unsigned short* p = Wk_bf + (nt * 16 + lm) * DDIM + ks * 32 + lh * 16;
            uint4 lo = *(const uint4*)(p);
            uint4 hi = *(const uint4*)(p + 8);
            u32x8 tt = {lo.x, lo.y, lo.z, lo.w, hi.x, hi.y, hi.z, hi.w};
            bk[ks] = tt;
        }
        #pragma unroll
        for (int mt = 0; mt < 2; ++mt) {
            v8f c = {0.f, 0.f, 0.f, 0.f, 0.f, 0.f, 0.f, 0.f};
            c = wmma_bf16(axn[mt][0], bk[0], c);
            c = wmma_bf16(axn[mt][1], bk[1], c);
            int colg = nt * 16 + lm;
            #pragma unroll
            for (int r = 0; r < 8; ++r)
                s_kh[(mt * 16 + lh * 8 + r) * (DDIM * HH) + colg] = f2bf(c[r]);
        }
    }

    // ---- Vsum projection: vsum[n,i] = xn @ Wvsum^T ; one (mt,nt) pair per wave ----
    {
        int mt = w >> 2, nt = w & 3;
        u32x8 bv[2];
        #pragma unroll
        for (int ks = 0; ks < 2; ++ks) {
            const float* p = Wvsum + (nt * 16 + lm) * DDIM + ks * 32 + lh * 16;
            unsigned dw[8];
            #pragma unroll
            for (int q = 0; q < 8; ++q)
                dw[q] = (unsigned)f2bf(p[2 * q]) | ((unsigned)f2bf(p[2 * q + 1]) << 16);
            u32x8 tt = {dw[0], dw[1], dw[2], dw[3], dw[4], dw[5], dw[6], dw[7]};
            bv[ks] = tt;
        }
        v8f c = {0.f, 0.f, 0.f, 0.f, 0.f, 0.f, 0.f, 0.f};
        c = wmma_bf16(axn[mt][0], bv[0], c);
        c = wmma_bf16(axn[mt][1], bv[1], c);
        #pragma unroll
        for (int r = 0; r < 8; ++r)
            s_vsum[(mt * 16 + lh * 8 + r) * DDIM + nt * 16 + lm] = c[r];
    }
    __syncthreads();

    // ---- scores = Qh @ Kh_n^T (K=8 zero-padded to 32 in registers), softmax over i,
    //      then acc[j] += sum_i P[i,j]*vsum[n,i].  Wave w owns n = w, w+8, w+16, w+24.
    u32x8 aq[4];
    #pragma unroll
    for (int mi = 0; mi < 4; ++mi) {
        u32x8 z = {0, 0, 0, 0, 0, 0, 0, 0};
        if (lh == 0) {
            const uint4 lo = *(const uint4*)(s_qh + (mi * 16 + lm) * HH);
            z[0] = lo.x; z[1] = lo.y; z[2] = lo.z; z[3] = lo.w;
        }
        aq[mi] = z;
    }

    #pragma unroll 1
    for (int nn = 0; nn < 4; ++nn) {
        int n = w + nn * 8;
        u32x8 bk[4];
        #pragma unroll
        for (int nj = 0; nj < 4; ++nj) {
            u32x8 z = {0, 0, 0, 0, 0, 0, 0, 0};
            if (lh == 0) {
                const uint4 lo = *(const uint4*)(s_kh + n * (DDIM * HH) + (nj * 16 + lm) * HH);
                z[0] = lo.x; z[1] = lo.y; z[2] = lo.z; z[3] = lo.w;
            }
            bk[nj] = z;
        }
        v8f c[4][4];
        #pragma unroll
        for (int mi = 0; mi < 4; ++mi)
            #pragma unroll
            for (int nj = 0; nj < 4; ++nj) {
                v8f z = {0.f, 0.f, 0.f, 0.f, 0.f, 0.f, 0.f, 0.f};
                c[mi][nj] = wmma_bf16(aq[mi], bk[nj], z);
            }

        float vsv[32];
        #pragma unroll
        for (int mi = 0; mi < 4; ++mi) {
            const float4 v0 = *(const float4*)(s_vsum + n * DDIM + mi * 16 + lh * 8);
            const float4 v1 = *(const float4*)(s_vsum + n * DDIM + mi * 16 + lh * 8 + 4);
            vsv[mi * 8 + 0] = v0.x; vsv[mi * 8 + 1] = v0.y;
            vsv[mi * 8 + 2] = v0.z; vsv[mi * 8 + 3] = v0.w;
            vsv[mi * 8 + 4] = v1.x; vsv[mi * 8 + 5] = v1.y;
            vsv[mi * 8 + 6] = v1.z; vsv[mi * 8 + 7] = v1.w;
        }

        #pragma unroll
        for (int nj = 0; nj < 4; ++nj) {
            float mx = -3.402823e38f;
            #pragma unroll
            for (int mi = 0; mi < 4; ++mi)
                #pragma unroll
                for (int r = 0; r < 8; ++r) mx = fmaxf(mx, c[mi][nj][r]);
            mx = fmaxf(mx, __shfl_xor(mx, 16, 32));   // combine M halves across lane^16
            float sum = 0.f, acc = 0.f;
            #pragma unroll
            for (int mi = 0; mi < 4; ++mi)
                #pragma unroll
                for (int r = 0; r < 8; ++r) {
                    float e = __expf(c[mi][nj][r] - mx);
                    sum += e;
                    acc += e * vsv[mi * 8 + r];
                }
            sum += __shfl_xor(sum, 16, 32);
            atomicAdd(&s_attn[nj * 16 + lm], acc / sum);  // both halves add partials
        }
    }
    __syncthreads();

    // ---- epilogue: h = attn(mean over H) @ Wo^T + bo + center ----
    if (tid < DDIM) {
        int j = tid;
        float acc = bo[j];
        #pragma unroll
        for (int k = 0; k < DDIM; ++k) acc += 0.125f * s_attn[k] * Wo[j * DDIM + k];
        h_pre[b * DDIM + j] = acc + s_center[j];
    }
}

// ---------------- BatchNorm statistics: one block per column ---------------
__global__ __launch_bounds__(256)
void bnstats_kernel(const float* __restrict__ x, float* __restrict__ mean,
                    float* __restrict__ rstd) {
    __shared__ float s_s[256], s_q[256];
    int j = blockIdx.x, tid = threadIdx.x;
    float s = 0.f, q = 0.f;
    for (int r = tid; r < BATCH; r += 256) {
        float v = x[r * DDIM + j];
        s += v; q += v * v;
    }
    s_s[tid] = s; s_q[tid] = q;
    __syncthreads();
    for (int off = 128; off > 0; off >>= 1) {
        if (tid < off) { s_s[tid] += s_s[tid + off]; s_q[tid] += s_q[tid + off]; }
        __syncthreads();
    }
    if (tid == 0) {
        float m = s_s[0] * (1.f / BATCH);
        float var = s_q[0] * (1.f / BATCH) - m * m;
        mean[j] = m;
        rstd[j] = rsqrtf(var + 1e-5f);
    }
}

// ---------------- BN1-apply + FFN + residual (4 rows per block) ------------
__global__ __launch_bounds__(256)
void ffn_kernel(const float* __restrict__ h_pre, const float* __restrict__ mean1,
                const float* __restrict__ rstd1, const float* __restrict__ gamma1,
                const float* __restrict__ beta1, const float* __restrict__ W1,
                const float* __restrict__ bf1, const float* __restrict__ W2,
                const float* __restrict__ bf2, float* __restrict__ y_pre) {
    __shared__ float s_h[4][DDIM];
    __shared__ float s_t[4][2 * DDIM];
    int tid = threadIdx.x;
    int rr = tid >> 6, j = tid & 63;
    int b = blockIdx.x * 4 + rr;
    float hn = (h_pre[b * DDIM + j] - mean1[j]) * rstd1[j] * gamma1[j] + beta1[j];
    s_h[rr][j] = hn;
    __syncthreads();
    #pragma unroll
    for (int kk = 0; kk < 2; ++kk) {
        int k = j + kk * 64;
        float acc = bf1[k];
        #pragma unroll
        for (int d = 0; d < DDIM; ++d) acc += s_h[rr][d] * W1[k * DDIM + d];
        s_t[rr][k] = fmaxf(acc, 0.f);
    }
    __syncthreads();
    float acc = bf2[j];
    #pragma unroll
    for (int k = 0; k < 2 * DDIM; ++k) acc += s_t[rr][k] * W2[j * (2 * DDIM) + k];
    y_pre[b * DDIM + j] = hn + acc;
}

// ---------------- BN2 apply --------------------------------------------------
__global__ __launch_bounds__(256)
void bnapply_kernel(const float* __restrict__ y_pre, const float* __restrict__ mean2,
                    const float* __restrict__ rstd2, const float* __restrict__ gamma2,
                    const float* __restrict__ beta2, float* __restrict__ out) {
    int i = blockIdx.x * 256 + threadIdx.x;   // 131072 total
    int j = i & 63;
    out[i] = (y_pre[i] - mean2[j]) * rstd2[j] * gamma2[j] + beta2[j];
}

extern "C" void kernel_launch(void* const* d_in, const int* in_sizes, int n_in,
                              void* d_out, int out_size, void* d_ws, size_t ws_size,
                              hipStream_t stream) {
    const float* center   = (const float*)d_in[0];
    const float* neighbor = (const float*)d_in[1];
    const float* Wq       = (const float*)d_in[2];
    const float* Wk       = (const float*)d_in[3];
    const float* Wv       = (const float*)d_in[4];
    const float* Wo       = (const float*)d_in[5];
    const float* bo       = (const float*)d_in[6];
    const float* gamma1   = (const float*)d_in[7];
    const float* beta1    = (const float*)d_in[8];
    const float* W1       = (const float*)d_in[9];
    const float* bf1      = (const float*)d_in[10];
    const float* W2       = (const float*)d_in[11];
    const float* bf2      = (const float*)d_in[12];
    const float* gamma2   = (const float*)d_in[13];
    const float* beta2    = (const float*)d_in[14];
    (void)in_sizes; (void)n_in; (void)out_size; (void)ws_size;

    char* ws = (char*)d_ws;
    float*          Wvsum = (float*)(ws);                                   // 16 KB
    unsigned short* Wk_bf = (unsigned short*)(ws + 16384);                  // 64 KB
    float*          h_pre = (float*)(ws + 16384 + 65536);                   // 512 KB
    float*          y_pre = (float*)(ws + 16384 + 65536 + 524288);          // 512 KB
    float*          stats = (float*)(ws + 16384 + 65536 + 2 * 524288);      // 1 KB
    float* mean1 = stats, *rstd1 = stats + 64, *mean2 = stats + 128, *rstd2 = stats + 192;
    float* out = (float*)d_out;

    prep_kernel   <<<128, 256, 0, stream>>>(Wv, Wk, Wvsum, Wk_bf);
    attn_kernel   <<<BATCH, 256, 0, stream>>>(center, neighbor, Wq, Wk_bf, Wvsum, Wo, bo, h_pre);
    bnstats_kernel<<<DDIM, 256, 0, stream>>>(h_pre, mean1, rstd1);
    ffn_kernel    <<<BATCH / 4, 256, 0, stream>>>(h_pre, mean1, rstd1, gamma1, beta1,
                                                  W1, bf1, W2, bf2, y_pre);
    bnstats_kernel<<<DDIM, 256, 0, stream>>>(y_pre, mean2, rstd2);
    bnapply_kernel<<<BATCH * DDIM / 256, 256, 0, stream>>>(y_pre, mean2, rstd2, gamma2, beta2, out);
}
